// MoE_Bottleneck_19576460935691
// MI455X (gfx1250) — matrix operand order repaired and checked by reading, not compiled, not run on hardware
//
#include <hip/hip_runtime.h>

// ---------------- CDNA5 WMMA / TDM types ----------------
typedef __attribute__((ext_vector_type(16))) __bf16 v16bf;
typedef __attribute__((ext_vector_type(8)))  float  v8f;
typedef __attribute__((ext_vector_type(4)))  unsigned int u32x4;
typedef __attribute__((ext_vector_type(8)))  int i32x8;
typedef __attribute__((ext_vector_type(4)))  int i32x4;

#define HW_    6400
#define C1_    256
#define C2_    256
#define CH_    128
#define NTOK_  204800
#define BN_EPS 1e-3f

// ---------------- workspace layout (bytes) ----------------
#define OFF_TOK  0ull          // 204800*128 bf16
#define OFF_MOE  52428800ull   // 204800*128 bf16
#define OFF_WD   104857600ull  // 204800*4 f32
#define OFF_W1P  108134400ull  // 256x128 bf16 packed (64KB)
#define OFF_CP   108199936ull  // 4x128x128 bf16 packed (128KB)
#define OFF_W3P  108331008ull  // 128x256 bf16 packed (64KB)
#define OFF_S1   108396544ull
#define OFF_H1   108397056ull
#define OFF_SE   108397568ull
#define OFF_HE   108399616ull
#define OFF_S3   108401664ull
#define OFF_H3   108402688ull

static __device__ __forceinline__ unsigned short f2bf(float f) {  // exact RNE
  unsigned int u = __float_as_uint(f);
  u += 0x7fffu + ((u >> 16) & 1u);
  return (unsigned short)(u >> 16);
}
// packed f32x2 -> bf16x2, round-to-nearest: 2x v_add + 1x v_perm_b32
#if __has_builtin(__builtin_amdgcn_cvt_pk_bf16_f32)
static __device__ __forceinline__ unsigned int pk2bf(float a, float b) {
  auto r = __builtin_amdgcn_cvt_pk_bf16_f32(a, b);
  unsigned int u;
  __builtin_memcpy(&u, &r, 4);
  return u;
}
#elif __has_builtin(__builtin_amdgcn_perm)
static __device__ __forceinline__ unsigned int pk2bf(float a, float b) {
  unsigned int ua = __float_as_uint(a) + 0x8000u;
  unsigned int ub = __float_as_uint(b) + 0x8000u;
  return __builtin_amdgcn_perm(ub, ua, 0x07060302u);  // {b.hi16, a.hi16}
}
#else
static __device__ __forceinline__ unsigned int pk2bf(float a, float b) {
  return (unsigned int)f2bf(a) | ((unsigned int)f2bf(b) << 16);
}
#endif
// bf16 store: backend folds (u>>16) into global_store_d16_hi_b16 (1 VALU)
static __device__ __forceinline__ void st_bf16(unsigned short* p, float f) {
  unsigned int u = __float_as_uint(f) + 0x8000u;
  *p = (unsigned short)(u >> 16);
}
static __device__ __forceinline__ float sigf(float y) {
  return 1.0f / (1.0f + __expf(-y));
}
static __device__ __forceinline__ v16bf ld_b_frag(const unsigned short* p) {
  union { uint4 q[2]; v16bf v; } u;
  const uint4* q = reinterpret_cast<const uint4*>(p);
  u.q[0] = q[0]; u.q[1] = q[1];
  return u.v;
}
static __device__ __forceinline__ v16bf ld_a_frag(const unsigned short* p) {
  union { uint4 q[2]; v16bf v; } u;
  u.q[0] = *reinterpret_cast<const uint4*>(p);
  u.q[1] = *reinterpret_cast<const uint4*>(p + 16);
  return u.v;
}
static __device__ __forceinline__ v8f wmma_bf16(v16bf a, v16bf b, v8f c) {
  return __builtin_amdgcn_wmma_f32_16x16x32_bf16(false, a, false, b, (short)0, c,
                                                 false, false);
}

// ---------------- Tensor Data Mover (ISA ch.8 D# descriptor) ----------------
#if __has_builtin(__builtin_amdgcn_tensor_load_to_lds)
#define HAVE_TDM 1
static __device__ __forceinline__ void tdm_load(
    const void* gptr, unsigned lds_off, unsigned data_size_code,
    unsigned tile_d0, unsigned tile_d1,
    unsigned long long tensor_d0, unsigned long long tensor_d1,
    unsigned long long stride0, unsigned pad_bits) {
  unsigned long long ga = (unsigned long long)gptr;
  u32x4 g0;
  g0[0] = 1u;                                                   // count=1
  g0[1] = lds_off;                                              // lds_addr
  g0[2] = (unsigned)(ga & 0xffffffffu);                         // gaddr[31:0]
  g0[3] = (unsigned)((ga >> 32) & 0x1ffffffu) | (2u << 30);     // gaddr hi|type=2
  i32x8 g1;
  g1[0] = (int)((data_size_code << 16) | pad_bits);             // mask=0,ds,pad
  g1[1] = (int)((unsigned)(tensor_d0 & 0xffffu) << 16);         // dim0[15:0]
  g1[2] = (int)(((tensor_d0 >> 16) & 0xffffu) |
                ((tensor_d1 & 0xffffu) << 16));                 // dim0 hi,dim1 lo
  g1[3] = (int)(((tensor_d1 >> 16) & 0xffffu) |
                ((tile_d0 & 0xffffu) << 16));                   // dim1 hi,tile0
  g1[4] = (int)(tile_d1 & 0xffffu);                             // tile1, tile2=0
  g1[5] = (int)(stride0 & 0xffffffffu);
  g1[6] = (int)((stride0 >> 32) & 0xffffu);
  g1[7] = 0;
  i32x4 z = {0, 0, 0, 0};
#if defined(__clang_major__) && (__clang_major__ >= 23)
  i32x8 z8 = {0, 0, 0, 0, 0, 0, 0, 0};
  __builtin_amdgcn_tensor_load_to_lds(g0, g1, z, z, z8, 0);
#else
  __builtin_amdgcn_tensor_load_to_lds(g0, g1, z, z, 0);
#endif
}
static __device__ __forceinline__ unsigned lds_off_of(const void* p) {
  return (unsigned)(unsigned long long)(uintptr_t)p;   // flat low 32 = LDS offset
}
#endif

static __device__ __forceinline__ void coop_copy16(void* dst, const void* src,
                                                   int n16, int tid, int nthr) {
  uint4* d = (uint4*)dst;
  const uint4* s = (const uint4*)src;
  for (int i = tid; i < n16; i += nthr) d[i] = s[i];
}

// ---------------- prep: pack weights into WMMA B-fragment order + fold BN ----
__global__ __launch_bounds__(256) void k_prep(
    const float* __restrict__ w1, const float* __restrict__ ew,
    const float* __restrict__ w3,
    const float* g1, const float* b1, const float* m1, const float* v1,
    const float* ge, const float* be, const float* me, const float* ve,
    const float* g3, const float* b3, const float* m3, const float* v3,
    unsigned short* __restrict__ w1p, unsigned short* __restrict__ cp,
    unsigned short* __restrict__ w3p,
    float* s1, float* h1, float* se, float* he, float* s3, float* h3) {
  int idx = blockIdx.x * 256 + threadIdx.x;
  if (idx < 32768) {                       // w1^T : 8 kt x 8 nt
    int t = idx >> 9, r = idx & 511, lane = r >> 4, j = r & 15;
    int kt = t >> 3, nt = t & 7;
    int k = kt * 32 + (j & 7) + ((j >> 3) << 4) + ((lane >> 4) << 3);
    int o = nt * 16 + (lane & 15);
    w1p[idx] = f2bf(w1[o * C1_ + k]);
  } else if (idx < 98304) {                // center^T per expert: 4 kt x 8 nt
    int i = idx - 32768;
    int e = i >> 14, r0 = i & 16383;
    int t = r0 >> 9, r = r0 & 511, lane = r >> 4, j = r & 15;
    int kt = t >> 3, nt = t & 7;
    int k = kt * 32 + (j & 7) + ((j >> 3) << 4) + ((lane >> 4) << 3);
    int o = nt * 16 + (lane & 15);
    cp[i] = f2bf(ew[(((size_t)e * CH_ + o) * CH_ + k) * 9 + 4]); // [1][1] tap
  } else if (idx < 131072) {               // w3^T : 4 kt x 16 nt
    int i = idx - 98304;
    int t = i >> 9, r = i & 511, lane = r >> 4, j = r & 15;
    int kt = t >> 4, nt = t & 15;
    int k = kt * 32 + (j & 7) + ((j >> 3) << 4) + ((lane >> 4) << 3);
    int o = nt * 16 + (lane & 15);
    w3p[i] = f2bf(w3[o * CH_ + k]);
  } else if (idx < 131072 + 128) {
    int o = idx - 131072;
    float inv = g1[o] * rsqrtf(v1[o] + BN_EPS);
    s1[o] = inv; h1[o] = b1[o] - m1[o] * inv;
  } else if (idx < 131072 + 128 + 512) {
    int o = idx - 131072 - 128;
    float inv = ge[o] * rsqrtf(ve[o] + BN_EPS);
    se[o] = inv; he[o] = be[o] - me[o] * inv;
  } else if (idx < 131072 + 128 + 512 + 256) {
    int o = idx - 131072 - 128 - 512;
    float inv = g3[o] * rsqrtf(v3[o] + BN_EPS);
    s3[o] = inv; h3[o] = b3[o] - m3[o] * inv;
  }
}

// ---------------- conv1: tokens = silu(bn1(w1 @ x)), bf16 out ---------------
__global__ __launch_bounds__(128) void k_conv1(
    const float* __restrict__ x, const unsigned short* __restrict__ w1p,
    const float* __restrict__ s1, const float* __restrict__ h1,
    unsigned short* __restrict__ tok) {
  __shared__ float xs[256 * 65];            // 66,560 B (TDM-padded rows)
  __shared__ unsigned short w1s[32768];     // 65,536 B
  const int lane = threadIdx.x & 31;
  const int wave = threadIdx.x >> 5;
  const int blkTok = blockIdx.x * 64;       // 6400 % 64 == 0 -> same image b
  const int b = blkTok / HW_;
  const int hwBase = blkTok - b * HW_;
  const float* xg = x + (size_t)b * C1_ * HW_ + hwBase;

#ifdef HAVE_TDM
  if (threadIdx.x < 32) {                   // wave 0 issues both TDM descriptors
    // x tile: f32, 64/row, 256 rows, stride HW_, LDS pad 1 dword per 64
    tdm_load(xg, lds_off_of(xs), 2u, 64u, 256u, 64ull, 256ull,
             (unsigned long long)HW_, (1u << 20) | (5u << 22));
    tdm_load(w1p, lds_off_of(w1s), 3u, 8192u, 1u, 8192ull, 1ull, 8192ull, 0u);
    __builtin_amdgcn_s_wait_tensorcnt(0);
  }
#else
  for (int i = threadIdx.x; i < 256 * 16; i += 128) {
    int c = i >> 4, seg = (i & 15) * 4;
    float4 v4 = *reinterpret_cast<const float4*>(xg + (size_t)c * HW_ + seg);
    float* d = &xs[c * 65 + seg];
    d[0] = v4.x; d[1] = v4.y; d[2] = v4.z; d[3] = v4.w;
  }
  coop_copy16(w1s, w1p, 4096, threadIdx.x, 128);
#endif
  __syncthreads();

  const int half = lane >> 4;
  const int m = lane & 15;
  const int col = wave * 16 + m;
  const int tokBase = blkTok + wave * 16;

  v8f acc[8] = {};
  for (int kt = 0; kt < 8; ++kt) {
    const int k0 = kt * 32 + half * 8;
    union { unsigned int w[8]; v16bf v; } af;   // A fragment (ISA 7.12.2)
#pragma unroll
    for (int i = 0; i < 4; ++i)
      af.w[i] = pk2bf(xs[(k0 + 2 * i) * 65 + col],
                      xs[(k0 + 2 * i + 1) * 65 + col]);
#pragma unroll
    for (int i = 0; i < 4; ++i)
      af.w[4 + i] = pk2bf(xs[(k0 + 16 + 2 * i) * 65 + col],
                          xs[(k0 + 17 + 2 * i) * 65 + col]);
    const unsigned short* bb = w1s + (size_t)(kt * 8) * 512 + lane * 16;
#pragma unroll
    for (int nt = 0; nt < 8; ++nt)
      acc[nt] = wmma_bf16(af.v, ld_b_frag(bb + nt * 512), acc[nt]);
  }
  const int ol = lane & 15;
#pragma unroll
  for (int nt = 0; nt < 8; ++nt) {
    const int o = nt * 16 + ol;
    const float sc = s1[o], sh = h1[o];
#pragma unroll
    for (int i = 0; i < 8; ++i) {
      const int tk = tokBase + i + 8 * half;
      float y = acc[nt][i] * sc + sh;
      st_bf16(&tok[(size_t)tk * CH_ + o], y * sigf(y));
    }
  }
}

// ---------------- router: top-2 softmax dispatch weights --------------------
__global__ __launch_bounds__(256) void k_router(
    const unsigned short* __restrict__ tok, const float* __restrict__ rw_g,
    const float* __restrict__ rb, float* __restrict__ wd) {
  __shared__ float rw[512];
  const int t = threadIdx.x;
  rw[t] = rw_g[t];
  rw[t + 256] = rw_g[t + 256];
  __syncthreads();
  const int n = blockIdx.x * 256 + t;
  float lg[4] = {rb[0], rb[1], rb[2], rb[3]};
  const uint4* tp = reinterpret_cast<const uint4*>(tok + (size_t)n * CH_);
#pragma unroll 4
  for (int ch = 0; ch < 16; ++ch) {
    uint4 q = tp[ch];
    unsigned int w4[4] = {q.x, q.y, q.z, q.w};
#pragma unroll
    for (int j = 0; j < 4; ++j) {
      float a = __uint_as_float(w4[j] << 16);
      float c = __uint_as_float(w4[j] & 0xffff0000u);
      int k = ch * 8 + j * 2;
#pragma unroll
      for (int e = 0; e < 4; ++e)
        lg[e] += a * rw[e * 128 + k] + c * rw[e * 128 + k + 1];
    }
  }
  int i1 = 0;
#pragma unroll
  for (int e = 1; e < 4; ++e) if (lg[e] > lg[i1]) i1 = e;
  int i2 = (i1 == 0) ? 1 : 0;
#pragma unroll
  for (int e = 0; e < 4; ++e) if (e != i1 && lg[e] > lg[i2]) i2 = e;
  float p2 = __expf(lg[i2] - lg[i1]);       // softmax over top-2 (rest cancels)
  float wa = 1.0f / (1.0f + p2), wb = p2 * wa;
  float o4[4] = {0.f, 0.f, 0.f, 0.f};
  o4[i1] = wa; o4[i2] = wb;
  reinterpret_cast<float4*>(wd)[n] = make_float4(o4[0], o4[1], o4[2], o4[3]);
}

// ---------------- experts: gated sum of silu(bn_e(tok @ center_e^T)) --------
// TDM stages all 4 experts' packed B (128KB) + the 64-token A tile (16KB) +
// dispatch weights (1KB); GEMMs run entirely out of LDS.
__global__ __launch_bounds__(128) void k_moe(
    const unsigned short* __restrict__ tok, const unsigned short* __restrict__ cp,
    const float* __restrict__ se, const float* __restrict__ he,
    const float* __restrict__ wd, unsigned short* __restrict__ moe) {
  __shared__ unsigned short cps[65536];     // 128 KB
  __shared__ unsigned short toks[8192];     // 16 KB (64 tokens x 128)
  __shared__ float wds[256];                // 1 KB  (64 tokens x 4)
  const int blkTok = blockIdx.x * 64;
#ifdef HAVE_TDM
  if (threadIdx.x < 32) {
    tdm_load(cp, lds_off_of(cps), 3u, 16384u, 1u, 16384ull, 1ull, 16384ull, 0u);
    tdm_load(tok + (size_t)blkTok * CH_, lds_off_of(toks), 3u, 2048u, 1u,
             2048ull, 1ull, 2048ull, 0u);
    tdm_load(wd + (size_t)blkTok * 4, lds_off_of(wds), 3u, 128u, 1u,
             128ull, 1ull, 128ull, 0u);
    __builtin_amdgcn_s_wait_tensorcnt(0);
  }
#else
  coop_copy16(cps, cp, 8192, threadIdx.x, 128);
  coop_copy16(toks, tok + (size_t)blkTok * CH_, 1024, threadIdx.x, 128);
  coop_copy16(wds, wd + (size_t)blkTok * 4, 64, threadIdx.x, 128);
#endif
  __syncthreads();

  const int lane = threadIdx.x & 31;
  const int wave = threadIdx.x >> 5;
  const int tokBase = blkTok + wave * 16;
  const int half = lane >> 4;
  const int m = lane & 15;
  const unsigned short* arow = toks + (size_t)(wave * 16 + m) * CH_ + half * 8;
  const int ol = lane & 15;

  float macc[8][8] = {};
  for (int e = 0; e < 4; ++e) {
    v8f acc[8] = {};
#pragma unroll
    for (int kt = 0; kt < 4; ++kt) {
      v16bf a = ld_a_frag(arow + kt * 32);
      const unsigned short* bb =
          cps + (size_t)e * 16384 + (size_t)(kt * 8) * 512 + lane * 16;
#pragma unroll
      for (int nt = 0; nt < 8; ++nt)
        acc[nt] = wmma_bf16(a, ld_b_frag(bb + nt * 512), acc[nt]);
    }
    float wv[8];
#pragma unroll
    for (int i = 0; i < 8; ++i)
      wv[i] = wds[(wave * 16 + i + 8 * half) * 4 + e];
    const float* sep = se + e * CH_;
    const float* hep = he + e * CH_;
#pragma unroll
    for (int nt = 0; nt < 8; ++nt) {
      const int o = nt * 16 + ol;
      const float sc = sep[o], sh = hep[o];
#pragma unroll
      for (int i = 0; i < 8; ++i) {
        float y = acc[nt][i] * sc + sh;
        macc[nt][i] += wv[i] * (y * sigf(y));
      }
    }
  }
#pragma unroll
  for (int nt = 0; nt < 8; ++nt) {
    const int o = nt * 16 + ol;
#pragma unroll
    for (int i = 0; i < 8; ++i)
      st_bf16(&moe[(size_t)(tokBase + i + 8 * half) * CH_ + o], macc[nt][i]);
  }
}

// ---------------- conv3: out = silu(bn3(w3 @ moe)) + x ----------------------
__global__ __launch_bounds__(128) void k_conv3(
    const unsigned short* __restrict__ moe, const unsigned short* __restrict__ w3p,
    const float* __restrict__ s3, const float* __restrict__ h3,
    const float* __restrict__ x, float* __restrict__ out) {
  __shared__ unsigned short w3s[32768];     // 64 KB
  __shared__ unsigned short moes[8192];     // 16 KB (64 tokens x 128)
  const int blkTok = blockIdx.x * 64;
#ifdef HAVE_TDM
  if (threadIdx.x < 32) {
    tdm_load(w3p, lds_off_of(w3s), 3u, 8192u, 1u, 8192ull, 1ull, 8192ull, 0u);
    tdm_load(moe + (size_t)blkTok * CH_, lds_off_of(moes), 3u, 2048u, 1u,
             2048ull, 1ull, 2048ull, 0u);
    __builtin_amdgcn_s_wait_tensorcnt(0);
  }
#else
  coop_copy16(w3s, w3p, 4096, threadIdx.x, 128);
  coop_copy16(moes, moe + (size_t)blkTok * CH_, 1024, threadIdx.x, 128);
#endif
  __syncthreads();

  const int lane = threadIdx.x & 31;
  const int wave = threadIdx.x >> 5;
  const int tokBase = blkTok + wave * 16;
  const int half = lane >> 4;
  const int m = lane & 15;
  const unsigned short* arow = moes + (size_t)(wave * 16 + m) * CH_ + half * 8;
  const int b = tokBase / HW_;
  const int hwBase = tokBase - b * HW_;
  const int ol = lane & 15;

  v8f acc[8] = {};
#pragma unroll
  for (int kt = 0; kt < 4; ++kt) {
    v16bf a = ld_a_frag(arow + kt * 32);
#pragma unroll
    for (int nt = 0; nt < 8; ++nt) {
      const int ntg = blockIdx.y * 8 + nt;
      const unsigned short* bb = w3s + (size_t)(kt * 16 + ntg) * 512 + lane * 16;
      acc[nt] = wmma_bf16(a, ld_b_frag(bb), acc[nt]);
    }
  }
#pragma unroll
  for (int nt = 0; nt < 8; ++nt) {
    const int o = blockIdx.y * 128 + nt * 16 + ol;
    const float sc = s3[o], sh = h3[o];
    const float* xrow = x + ((size_t)b * C2_ + o) * HW_ + hwBase;
    float* orow = out + ((size_t)b * C2_ + o) * HW_ + hwBase;
#pragma unroll
    for (int i = 0; i < 8; ++i) {
      const int r = i + 8 * half;
      float y = acc[nt][i] * sc + sh;
      orow[r] = y * sigf(y) + xrow[r];
    }
  }
}

// ---------------- launch ----------------------------------------------------
extern "C" void kernel_launch(void* const* d_in, const int* in_sizes, int n_in,
                              void* d_out, int out_size, void* d_ws, size_t ws_size,
                              hipStream_t stream) {
  const float* x  = (const float*)d_in[0];
  const float* w1 = (const float*)d_in[1];
  const float* g1 = (const float*)d_in[2];
  const float* b1 = (const float*)d_in[3];
  const float* m1 = (const float*)d_in[4];
  const float* v1 = (const float*)d_in[5];
  const float* rw = (const float*)d_in[6];
  const float* rb = (const float*)d_in[7];
  const float* ew = (const float*)d_in[8];
  const float* ge = (const float*)d_in[9];
  const float* be = (const float*)d_in[10];
  const float* me = (const float*)d_in[11];
  const float* ve = (const float*)d_in[12];
  const float* w3 = (const float*)d_in[13];
  const float* g3 = (const float*)d_in[14];
  const float* b3 = (const float*)d_in[15];
  const float* m3 = (const float*)d_in[16];
  const float* v3 = (const float*)d_in[17];
  float* out = (float*)d_out;

  char* ws = (char*)d_ws;
  unsigned short* tok = (unsigned short*)(ws + OFF_TOK);
  unsigned short* moe = (unsigned short*)(ws + OFF_MOE);
  float* wd           = (float*)(ws + OFF_WD);
  unsigned short* w1p = (unsigned short*)(ws + OFF_W1P);
  unsigned short* cp  = (unsigned short*)(ws + OFF_CP);
  unsigned short* w3p = (unsigned short*)(ws + OFF_W3P);
  float* s1 = (float*)(ws + OFF_S1);
  float* h1 = (float*)(ws + OFF_H1);
  float* se = (float*)(ws + OFF_SE);
  float* he = (float*)(ws + OFF_HE);
  float* s3 = (float*)(ws + OFF_S3);
  float* h3 = (float*)(ws + OFF_H3);

  k_prep<<<516, 256, 0, stream>>>(w1, ew, w3, g1, b1, m1, v1, ge, be, me, ve,
                                  g3, b3, m3, v3, w1p, cp, w3p,
                                  s1, h1, se, he, s3, h3);
  k_conv1<<<NTOK_ / 64, 128, 0, stream>>>(x, w1p, s1, h1, tok);
  k_router<<<NTOK_ / 256, 256, 0, stream>>>(tok, rw, rb, wd);
  k_moe<<<NTOK_ / 64, 128, 0, stream>>>(tok, cp, se, he, wd, moe);
  k_conv3<<<dim3(NTOK_ / 64, 2), 128, 0, stream>>>(moe, w3p, s3, h3, x, out);
}
